// BiaffineAttention_39968965656812
// MI455X (gfx1250) — compile-verified, matrix operand order checked by reading, not run on hardware
//
#include <hip/hip_runtime.h>
#include <hip/hip_bf16.h>

// ---------------------------------------------------------------------------
// BiaffineAttention on gfx1250 (MI455X): split-bf16 WMMA + async-LDS pipeline.
//   P : hs, weights -> zero-padded split-bf16 planes (hi + lo correction)
//   G1: X1h = relu(hs @ w1h^T + b1h)            [8192 x 512]
//   G2: X1d = relu(hs @ w1d^T + b1d)
//   G3: headW = X1h @ W2'^T + b2'   (W2' = Wb^T w2h, b2' = b2h Wb  -- folds Wb)
//   G4: dep   = X1d @ w2d^T + b2d
//   G5: scores[b,i,j] = headW[b,i] . dep[b,j] + bb  -> d_out fp32
// All N/K dims padded 500 -> 512 with zeros so no masking is needed anywhere.
// GEMM: 256 thr (8 waves, 2x4), block tile 128x128, BK=32, double-buffered LDS
// filled by GLOBAL_LOAD_ASYNC_TO_LDS_B128 (ASYNCcnt), XOR-swizzled banks.
// ---------------------------------------------------------------------------

typedef __attribute__((ext_vector_type(16))) __bf16 v16bf;
typedef __attribute__((ext_vector_type(8)))  __bf16 v8bf;
typedef __attribute__((ext_vector_type(8)))  float  v8f;

#define BM     128
#define BN     128
#define BK     32
#define PLANE  8192u              // 128 rows * 64B per plane
#define BUFSZ  32768u             // 4 planes
#define LDSSZ  65536u             // 2 buffers

// Fragment for the WMMA 16-bit operand: lane m/n = lane&15, s8 = lane>>4.
// Logical 16B chunks {s8, s8+2} of the row, XOR-swizzled by (row>>2)&3.
__device__ __forceinline__ v16bf load_frag(const char* plane, int row, int s8) {
    int rx = (row >> 2) & 3;
    const char* rb = plane + row * 64;
    v8bf a = *(const v8bf*)(rb + ((s8 ^ rx) & 3) * 16);
    v8bf b = *(const v8bf*)(rb + (((s8 + 2) ^ rx) & 3) * 16);
    return __builtin_shufflevector(a, b, 0,1,2,3,4,5,6,7,8,9,10,11,12,13,14,15);
}

// OM: 0 = bias[n] (+optional relu) -> split-bf16 output planes (ld 512)
//     1 = bias[0] scalar -> fp32 scores output (per-batch 512x512)
template<int OM, bool RELU>
__global__ __launch_bounds__(256) void gemm_bf16x2_async(
    const __bf16* __restrict__ AhiG, const __bf16* __restrict__ AloG, int lda,
    const __bf16* __restrict__ BhiG, const __bf16* __restrict__ BloG, int ldb,
    const float*  __restrict__ bias,
    __bf16* __restrict__ Ohi, __bf16* __restrict__ Olo, float* __restrict__ Ofp,
    int K, int zRowsA, int zRowsB)
{
    extern __shared__ char smem[];

    const int tid  = threadIdx.x;
    const int lane = tid & 31;
    const int wave = tid >> 5;
    const int wm   = wave >> 2;   // 0..1 : 64-row half
    const int wn   = wave & 3;    // 0..3 : 32-col quarter
    const int aRow0 = blockIdx.z * zRowsA + blockIdx.y * BM;
    const int bRow0 = blockIdx.z * zRowsB + blockIdx.x * BN;
    const unsigned ldsBase = __builtin_amdgcn_groupstaticsize();  // dyn LDS offset

    // DMA one 128x32 x {Ahi,Alo,Bhi,Blo} tile into LDS buffer `buf`.
    // 8 async b128 per thread; per-wave ASYNCcnt += 8, completes in order.
    auto issue_tile = [&](int buf, int k0) {
        unsigned base = ldsBase + (unsigned)buf * BUFSZ;
        #pragma unroll
        for (int pl = 0; pl < 4; ++pl) {
            const __bf16* src = (pl == 0) ? AhiG : (pl == 1) ? AloG
                              : (pl == 2) ? BhiG : BloG;
            const int row0 = (pl < 2) ? aRow0 : bRow0;
            const int ld   = (pl < 2) ? lda   : ldb;
            #pragma unroll
            for (int i = 0; i < 2; ++i) {
                int c  = (tid << 1) | i;      // 0..511 chunks of 16B
                int r  = c >> 2;
                int kc = c & 3;
                const __bf16* gp = src + (size_t)(row0 + r) * ld + k0 + kc * 8;
                unsigned la = base + pl * PLANE
                            + (unsigned)(r * 64 + ((kc ^ (r >> 2)) & 3) * 16);
                asm volatile("global_load_async_to_lds_b128 %0, %1, off"
                             :: "v"(la), "v"(gp) : "memory");
            }
        }
    };

    v8f acc[4][2];
    #pragma unroll
    for (int i = 0; i < 4; ++i)
        #pragma unroll
        for (int j = 0; j < 2; ++j)
            #pragma unroll
            for (int e = 0; e < 8; ++e) acc[i][j][e] = 0.0f;

    const int nk = K >> 5;
    issue_tile(0, 0);

    for (int it = 0; it < nk; ++it) {
        const int cur = it & 1;
        if (it + 1 < nk) {
            issue_tile(1 - cur, (it + 1) << 5);          // prefetch next slab
            asm volatile("s_wait_asynccnt 0x8" ::: "memory");  // slab `it` landed
        } else {
            asm volatile("s_wait_asynccnt 0x0" ::: "memory");
        }
        __syncthreads();   // every wave's DMA for slab `it` visible

        const char* base = smem + cur * BUFSZ;
        const char* pAhi = base;
        const char* pAlo = base + PLANE;
        const char* pBhi = base + 2 * PLANE;
        const char* pBlo = base + 3 * PLANE;

        const int s8 = lane >> 4;
        const int fr = lane & 15;
        v16bf aH[4], aL[4], bH[2], bL[2];
        #pragma unroll
        for (int t = 0; t < 4; ++t) {
            int r = wm * 64 + t * 16 + fr;
            aH[t] = load_frag(pAhi, r, s8);
            aL[t] = load_frag(pAlo, r, s8);
        }
        #pragma unroll
        for (int t = 0; t < 2; ++t) {
            int r = wn * 32 + t * 16 + fr;
            bH[t] = load_frag(pBhi, r, s8);
            bL[t] = load_frag(pBlo, r, s8);
        }
        #pragma unroll
        for (int i = 0; i < 4; ++i)
            #pragma unroll
            for (int j = 0; j < 2; ++j) {
                acc[i][j] = __builtin_amdgcn_wmma_f32_16x16x32_bf16(
                    false, aH[i], false, bH[j], (short)0, acc[i][j], false, false);
                acc[i][j] = __builtin_amdgcn_wmma_f32_16x16x32_bf16(
                    false, aH[i], false, bL[j], (short)0, acc[i][j], false, false);
                acc[i][j] = __builtin_amdgcn_wmma_f32_16x16x32_bf16(
                    false, aL[i], false, bH[j], (short)0, acc[i][j], false, false);
            }
        __syncthreads();   // all waves done reading buf `cur` -> reusable
    }

    // ---------------- epilogue ----------------
    const int rowAdd = (lane >> 4) << 3;   // C layout: lanes 16-31 hold M+8
    const int colIn  = lane & 15;
    #pragma unroll
    for (int i = 0; i < 4; ++i) {
        #pragma unroll
        for (int j = 0; j < 2; ++j) {
            int n = blockIdx.x * BN + wn * 32 + j * 16 + colIn;
            #pragma unroll
            for (int e = 0; e < 8; ++e) {
                int m = blockIdx.y * BM + wm * 64 + i * 16 + e + rowAdd;
                float v = acc[i][j][e];
                if (OM == 0) {
                    v += bias[n];
                    if (RELU) v = v > 0.0f ? v : 0.0f;
                    __bf16 h = (__bf16)v;
                    size_t o = (size_t)m * 512 + n;
                    Ohi[o] = h;
                    Olo[o] = (__bf16)(v - (float)h);
                } else {
                    v += bias[0];
                    Ofp[(size_t)blockIdx.z * (512 * 512) + (size_t)m * 512 + n] = v;
                }
            }
        }
    }
}

// ---------------- prep kernels (bandwidth-trivial) ----------------
// fp32 [ri x ci] -> zero-padded split-bf16 planes [ro x co]
__global__ void split_pad_kernel(const float* __restrict__ in,
                                 __bf16* __restrict__ oh, __bf16* __restrict__ ol,
                                 int ri, int ci, int ro, int co) {
    int idx = blockIdx.x * blockDim.x + threadIdx.x;
    if (idx >= ro * co) return;
    int r = idx / co, c = idx % co;
    float v = (r < ri && c < ci) ? in[(size_t)r * ci + c] : 0.0f;
    __bf16 h = (__bf16)v;
    oh[idx] = h;
    ol[idx] = (__bf16)(v - (float)h);
}

__global__ void pad_fp32_kernel(const float* __restrict__ in, float* __restrict__ out,
                                int ri, int ci, int ro, int co) {
    int idx = blockIdx.x * blockDim.x + threadIdx.x;
    if (idx >= ro * co) return;
    int r = idx / co, c = idx % co;
    out[idx] = (r < ri && c < ci) ? in[(size_t)r * ci + c] : 0.0f;
}

// W2'[n,j] = sum_a Wb[a,n] * w2h[a,j]   (padded 512x512 fp32)
__global__ void w2prime_kernel(const float* __restrict__ Wb, const float* __restrict__ w2h,
                               float* __restrict__ W2p) {
    int n = blockIdx.x;
    int j = blockIdx.y * blockDim.x + threadIdx.x;
    float s = 0.0f;
    if (n < 500 && j < 500)
        for (int a = 0; a < 500; ++a) s += Wb[a * 500 + n] * w2h[a * 500 + j];
    W2p[n * 512 + j] = s;
}

// b2'[n] = sum_a b2h[a] * Wb[a,n]       (padded to 512)
__global__ void b2prime_kernel(const float* __restrict__ Wb, const float* __restrict__ b2h,
                               float* __restrict__ b2p) {
    int n = blockIdx.x * blockDim.x + threadIdx.x;
    if (n >= 512) return;
    float s = 0.0f;
    if (n < 500)
        for (int a = 0; a < 500; ++a) s += b2h[a] * Wb[a * 500 + n];
    b2p[n] = s;
}

extern "C" void kernel_launch(void* const* d_in, const int* in_sizes, int n_in,
                              void* d_out, int out_size, void* d_ws, size_t ws_size,
                              hipStream_t stream) {
    const float* hs  = (const float*)d_in[0];
    const float* w1h = (const float*)d_in[1];
    const float* b1h = (const float*)d_in[2];
    const float* w2h = (const float*)d_in[3];
    const float* b2h = (const float*)d_in[4];
    const float* w1d = (const float*)d_in[5];
    const float* b1d = (const float*)d_in[6];
    const float* w2d = (const float*)d_in[7];
    const float* b2d = (const float*)d_in[8];
    const float* Wb  = (const float*)d_in[9];
    const float* bb  = (const float*)d_in[10];
    float* out = (float*)d_out;

    char* ws = (char*)d_ws;
    size_t off = 0;
    auto take = [&](size_t bytes) -> char* {
        char* p = ws + off;
        off += (bytes + 255) & ~(size_t)255;
        return p;
    };
    __bf16* hsH   = (__bf16*)take(8192ull * 1024 * 2);
    __bf16* hsL   = (__bf16*)take(8192ull * 1024 * 2);
    __bf16* w1hH  = (__bf16*)take(512ull * 1024 * 2);
    __bf16* w1hL  = (__bf16*)take(512ull * 1024 * 2);
    __bf16* w1dH  = (__bf16*)take(512ull * 1024 * 2);
    __bf16* w1dL  = (__bf16*)take(512ull * 1024 * 2);
    __bf16* w2dH  = (__bf16*)take(512ull * 512 * 2);
    __bf16* w2dL  = (__bf16*)take(512ull * 512 * 2);
    float*  W2p   = (float*)take(512ull * 512 * 4);
    __bf16* W2pH  = (__bf16*)take(512ull * 512 * 2);
    __bf16* W2pL  = (__bf16*)take(512ull * 512 * 2);
    float*  b1h_p = (float*)take(512 * 4);
    float*  b1d_p = (float*)take(512 * 4);
    float*  b2d_p = (float*)take(512 * 4);
    float*  b2p   = (float*)take(512 * 4);
    __bf16* x1hH  = (__bf16*)take(8192ull * 512 * 2);
    __bf16* x1hL  = (__bf16*)take(8192ull * 512 * 2);
    __bf16* x1dH  = (__bf16*)take(8192ull * 512 * 2);
    __bf16* x1dL  = (__bf16*)take(8192ull * 512 * 2);
    __bf16* hwH   = hsH;    // hs split dead after G2; reuse for headW
    __bf16* hwL   = hsL;
    __bf16* depH  = x1hH;   // X1h dead after G3; reuse for dep
    __bf16* depL  = x1hL;

    // ---- prep: split/pad operands, fold Wb into the second head GEMM ----
    split_pad_kernel<<<(8192 * 1024 + 255) / 256, 256, 0, stream>>>(hs, hsH, hsL, 8192, 1024, 8192, 1024);
    split_pad_kernel<<<(512 * 1024 + 255) / 256, 256, 0, stream>>>(w1h, w1hH, w1hL, 500, 1024, 512, 1024);
    split_pad_kernel<<<(512 * 1024 + 255) / 256, 256, 0, stream>>>(w1d, w1dH, w1dL, 500, 1024, 512, 1024);
    split_pad_kernel<<<(512 * 512 + 255) / 256, 256, 0, stream>>>(w2d, w2dH, w2dL, 500, 500, 512, 512);
    w2prime_kernel<<<dim3(512, 2), 256, 0, stream>>>(Wb, w2h, W2p);
    split_pad_kernel<<<(512 * 512 + 255) / 256, 256, 0, stream>>>(W2p, W2pH, W2pL, 512, 512, 512, 512);
    pad_fp32_kernel<<<2, 256, 0, stream>>>(b1h, b1h_p, 1, 500, 1, 512);
    pad_fp32_kernel<<<2, 256, 0, stream>>>(b1d, b1d_p, 1, 500, 1, 512);
    pad_fp32_kernel<<<2, 256, 0, stream>>>(b2d, b2d_p, 1, 500, 1, 512);
    b2prime_kernel<<<2, 256, 0, stream>>>(Wb, b2h, b2p);

    dim3 blk(256);
    dim3 g14(512 / BN, 8192 / BM, 1);

    // G1: X1h = relu(hs @ w1h^T + b1h)
    gemm_bf16x2_async<0, true><<<g14, blk, LDSSZ, stream>>>(
        hsH, hsL, 1024, w1hH, w1hL, 1024, b1h_p, x1hH, x1hL, nullptr, 1024, 0, 0);
    // G2: X1d = relu(hs @ w1d^T + b1d)
    gemm_bf16x2_async<0, true><<<g14, blk, LDSSZ, stream>>>(
        hsH, hsL, 1024, w1dH, w1dL, 1024, b1d_p, x1dH, x1dL, nullptr, 1024, 0, 0);
    // G3: headW = X1h @ W2'^T + b2'   (writes into reused hs planes)
    gemm_bf16x2_async<0, false><<<g14, blk, LDSSZ, stream>>>(
        x1hH, x1hL, 512, W2pH, W2pL, 512, b2p, hwH, hwL, nullptr, 512, 0, 0);
    // G4: dep = X1d @ w2d^T + b2d     (writes into reused X1h planes)
    gemm_bf16x2_async<0, false><<<g14, blk, LDSSZ, stream>>>(
        x1dH, x1dL, 512, w2dH, w2dL, 512, b2d_p, depH, depL, nullptr, 512, 0, 0);
    // G5: scores[b,i,j] = headW[b,i] . dep[b,j] + bb
    dim3 g5(512 / BN, 512 / BM, 16);
    gemm_bf16x2_async<1, false><<<g5, blk, LDSSZ, stream>>>(
        hwH, hwL, 512, depH, depL, 512, bb, nullptr, nullptr, out, 512, 512, 512);
}
